// DistanceLayer_4166118277258
// MI455X (gfx1250) — matrix operand order repaired and checked by reading, not compiled, not run on hardware
//
#include <hip/hip_runtime.h>
#include <hip/hip_bf16.h>

typedef __attribute__((ext_vector_type(16))) _Float16 v16h;
typedef __attribute__((ext_vector_type(8)))  float    v8f;

constexpr int NB      = 512;   // samples
constexpr int NT      = 2048;  // time steps
constexpr int NS      = 128;   // shapelets
constexpr int NL      = 64;    // shapelet length
constexpr int NP      = NT - NL + 1;   // 1985 windows
constexpr int PTILES  = (NP + 15) / 16; // 125 position tiles
constexpr int STILES  = 4;     // 16-shapelet tiles per workgroup (64 shapelets)
constexpr int NTHREADS = 256;
constexpr int NWAVES   = NTHREADS / 32;

union V16U { v16h v; unsigned u[8]; _Float16 h[16]; };

__global__ __launch_bounds__(NTHREADS)
void shapelet_dist_kernel(const float* __restrict__ x,
                          const float* __restrict__ shp,
                          float* __restrict__ out)
{
    __shared__ float    pre1[NT + 1];     // prefix sum of x
    __shared__ float    pre2[NT + 1];     // prefix sum of x^2
    __shared__ float    mup[2000];        // window means
    __shared__ float    p2c[2000];        // centered ||patch||^2
    __shared__ float    s2a[64], ssa[64]; // shapelet ||s||^2, sum(s)
    __shared__ _Float16 xhi[2068], xlo[2068];     // f16 split of x row (padded)
    __shared__ unsigned ph[2064], pl[2064];       // pair-packed {x[i],x[i+1]}
    __shared__ float    totx[NTHREADS], totx2[NTHREADS];
    __shared__ float    partial[NWAVES][64];

    const int tid   = threadIdx.x;
    const int b     = blockIdx.x;
    const int sbase = blockIdx.y * 64;
    const float* gx = x + (size_t)b * NT;

    // ---- pass 1: load x row, f16 hi/lo split, per-thread prefix of 8 elems ----
    float px[8], px2[8];
    {
        float run = 0.f, run2 = 0.f;
        #pragma unroll
        for (int j = 0; j < 8; ++j) {
            float v = gx[tid * 8 + j];
            _Float16 h = (_Float16)v;
            xhi[tid * 8 + j] = h;
            xlo[tid * 8 + j] = (_Float16)(v - (float)h);
            run  += v;     px[j]  = run;
            run2 += v * v; px2[j] = run2;
        }
        totx[tid] = run; totx2[tid] = run2;
    }
    for (int i = NT + tid; i < 2068; i += NTHREADS) {
        xhi[i] = (_Float16)0.f; xlo[i] = (_Float16)0.f;
    }
    __syncthreads();

    // ---- Hillis-Steele scan of 256 block totals ----
    float myTot = totx[tid], myTot2 = totx2[tid];
    for (int off = 1; off < NTHREADS; off <<= 1) {
        float a = 0.f, a2 = 0.f;
        if (tid >= off) { a = totx[tid - off]; a2 = totx2[tid - off]; }
        __syncthreads();
        if (tid >= off) { totx[tid] += a; totx2[tid] += a2; }
        __syncthreads();
    }
    float ex  = totx[tid]  - myTot;    // exclusive block prefix
    float ex2 = totx2[tid] - myTot2;
    #pragma unroll
    for (int j = 0; j < 8; ++j) {
        pre1[tid * 8 + j + 1] = ex  + px[j];
        pre2[tid * 8 + j + 1] = ex2 + px2[j];
    }
    if (tid == 0) { pre1[0] = 0.f; pre2[0] = 0.f; }
    __syncthreads();

    // ---- pair-packed arrays: any 8-run readable as 4 aligned ds_load_b32 ----
    for (int i = tid; i < 2064; i += NTHREADS) {
        union { _Float16 h[2]; unsigned u; } pk;
        pk.h[0] = xhi[i]; pk.h[1] = xhi[i + 1]; ph[i] = pk.u;
        pk.h[0] = xlo[i]; pk.h[1] = xlo[i + 1]; pl[i] = pk.u;
    }
    // ---- per-window mean and centered sum of squares ----
    for (int p = tid; p < 2000; p += NTHREADS) {
        if (p < NP) {
            float s  = pre1[p + NL] - pre1[p];
            float s2 = pre2[p + NL] - pre2[p];
            float mu = s * (1.f / NL);
            mup[p] = mu;
            p2c[p] = s2 - (float)NL * mu * mu;
        } else { mup[p] = 0.f; p2c[p] = 0.f; }
    }
    // ---- shapelet norms (exact fp32) for this WG's 64 shapelets ----
    if (tid < 64) {
        const float* sr = shp + (size_t)(sbase + tid) * NL;
        float ss = 0.f, s2 = 0.f;
        #pragma unroll 8
        for (int l = 0; l < NL; ++l) { float v = sr[l]; ss += v; s2 += v * v; }
        ssa[tid] = ss; s2a[tid] = s2;
    }
    __syncthreads();

    // ---- WMMA phase ----
    const int wave = tid >> 5;
    const int lane = tid & 31;
    const int n    = lane & 15;
    const int half = lane >> 4;

    // B = shapelets (K x N), register resident: lane n holds k = 8h..8h+7 and 16+8h..16+8h+7
    V16U Bh[STILES][2], Bl[STILES][2];
    #pragma unroll
    for (int t = 0; t < STILES; ++t) {
        #pragma unroll
        for (int kc = 0; kc < 2; ++kc) {
            const float4* sp = (const float4*)(shp +
                (size_t)(sbase + t * 16 + n) * NL + kc * 32 + half * 8);
            #pragma unroll
            for (int q = 0; q < 4; ++q) {
                float4 f = sp[q + (q & 2)];
                float e0 = f.x, e1 = f.y, e2 = f.z, e3 = f.w;
                _Float16 h0 = (_Float16)e0, h1 = (_Float16)e1;
                _Float16 h2 = (_Float16)e2, h3 = (_Float16)e3;
                Bh[t][kc].h[4*q+0] = h0; Bh[t][kc].h[4*q+1] = h1;
                Bh[t][kc].h[4*q+2] = h2; Bh[t][kc].h[4*q+3] = h3;
                Bl[t][kc].h[4*q+0] = (_Float16)(e0 - (float)h0);
                Bl[t][kc].h[4*q+1] = (_Float16)(e1 - (float)h1);
                Bl[t][kc].h[4*q+2] = (_Float16)(e2 - (float)h2);
                Bl[t][kc].h[4*q+3] = (_Float16)(e3 - (float)h3);
            }
        }
    }
    float s2v[STILES], ssv[STILES];
    #pragma unroll
    for (int t = 0; t < STILES; ++t) { s2v[t] = s2a[t*16 + n]; ssv[t] = ssa[t*16 + n]; }

    float rmin[STILES];
    #pragma unroll
    for (int t = 0; t < STILES; ++t) rmin[t] = 3.0e38f;

    for (int it = wave; it < PTILES; it += NWAVES) {
        const int p0 = it * 16;
        // A = 16x32 patch tile from pair-packed LDS (hi and lo)
        V16U Ah[2], Al[2];
        #pragma unroll
        for (int kc = 0; kc < 2; ++kc) {
            int base = p0 + n + kc * 32 + half * 8;
            #pragma unroll
            for (int q = 0; q < 4; ++q) {
                Ah[kc].u[q]     = ph[base + 2*q];
                Ah[kc].u[4 + q] = ph[base + 16 + 2*q];
                Al[kc].u[q]     = pl[base + 2*q];
                Al[kc].u[4 + q] = pl[base + 16 + 2*q];
            }
        }
        float muv[8], p2v[8];
        #pragma unroll
        for (int r = 0; r < 8; ++r) {
            int p = p0 + r + 8 * half;
            muv[r] = mup[p]; p2v[r] = p2c[p];
        }
        #pragma unroll
        for (int t = 0; t < STILES; ++t) {
            v8f acc = {};
            // split product: hi*hi + hi*lo + lo*hi  (~fp32 accuracy)
            acc = __builtin_amdgcn_wmma_f32_16x16x32_f16(false, Ah[0].v, false, Bh[t][0].v, (short)0, acc, false, false);
            acc = __builtin_amdgcn_wmma_f32_16x16x32_f16(false, Ah[0].v, false, Bl[t][0].v, (short)0, acc, false, false);
            acc = __builtin_amdgcn_wmma_f32_16x16x32_f16(false, Al[0].v, false, Bh[t][0].v, (short)0, acc, false, false);
            acc = __builtin_amdgcn_wmma_f32_16x16x32_f16(false, Ah[1].v, false, Bh[t][1].v, (short)0, acc, false, false);
            acc = __builtin_amdgcn_wmma_f32_16x16x32_f16(false, Ah[1].v, false, Bl[t][1].v, (short)0, acc, false, false);
            acc = __builtin_amdgcn_wmma_f32_16x16x32_f16(false, Al[1].v, false, Bh[t][1].v, (short)0, acc, false, false);
            #pragma unroll
            for (int r = 0; r < 8; ++r) {
                int p = p0 + r + 8 * half;
                float d2 = p2v[r] + s2v[t] - 2.f * acc[r] + 2.f * muv[r] * ssv[t];
                d2 = (p < NP) ? d2 : 3.0e38f;
                rmin[t] = fminf(rmin[t], d2);
            }
        }
    }

    // ---- reductions: lane halves -> waves -> output ----
    #pragma unroll
    for (int t = 0; t < STILES; ++t) {
        float o = __shfl_xor(rmin[t], 16, 32);
        rmin[t] = fminf(rmin[t], o);
        if (half == 0) partial[wave][t * 16 + n] = rmin[t];
    }
    __syncthreads();
    if (tid < 64) {
        float m = partial[0][tid];
        #pragma unroll
        for (int w = 1; w < NWAVES; ++w) m = fminf(m, partial[w][tid]);
        out[(size_t)b * NS + sbase + tid] = sqrtf(fmaxf(m, 0.f));
    }
}

extern "C" void kernel_launch(void* const* d_in, const int* in_sizes, int n_in,
                              void* d_out, int out_size, void* d_ws, size_t ws_size,
                              hipStream_t stream) {
    const float* x   = (const float*)d_in[0];
    const float* shp = (const float*)d_in[1];
    float* out = (float*)d_out;
    dim3 grid(NB, NS / (16 * STILES));  // (512, 2)
    shapelet_dist_kernel<<<grid, NTHREADS, 0, stream>>>(x, shp, out);
}
